// GraphEncoder_13572096655651
// MI455X (gfx1250) — matrix operand order repaired and checked by reading, not compiled, not run on hardware
//
#include <hip/hip_runtime.h>
#include <math.h>

// Problem constants (from reference)
#define Bn    64
#define Nn    512
#define F_IN  256
#define Dd    128
#define TOPK  20
#define HID   512

#define TILE_M 128     // rows per workgroup (8 waves x 16)
#define TILE_N 64      // cols per workgroup (4 x 16 per wave, B panel shared)
#define PANWB  40      // padded LDS bf16 panel row stride (20 dwords -> conflict-free)

typedef __attribute__((ext_vector_type(16))) __bf16 v16bf;
typedef __attribute__((ext_vector_type(8)))  __bf16 v8bf;
typedef __attribute__((ext_vector_type(8)))  float  v8f;
typedef __attribute__((ext_vector_type(4)))  int    v4i;

#if defined(__HIP_DEVICE_COMPILE__) && __has_builtin(__builtin_amdgcn_global_load_async_to_lds_b128)
#define HAVE_ASYNC_LDS 1
#else
#define HAVE_ASYNC_LDS 0
#endif

__device__ __forceinline__ void wait_async0() {
#if defined(__HIP_DEVICE_COMPILE__)
#if __has_builtin(__builtin_amdgcn_s_wait_asynccnt)
    __builtin_amdgcn_s_wait_asynccnt(0);
#else
    asm volatile("s_wait_asynccnt 0" ::: "memory");
#endif
#endif
}

// ---------------- fp32 -> bf16 conversion (one-time, weights) ----------------
__global__ void f2bf_k(const float* __restrict__ in, __bf16* __restrict__ out, int n) {
    int i = blockIdx.x * blockDim.x + threadIdx.x;
    if (i < n) out[i] = (__bf16)in[i];
}

// ---------------- 1) row norms of emb_weight ----------------
__global__ void row_norms_k(const float* __restrict__ W, float* __restrict__ nrm) {
    int n = blockIdx.x * blockDim.x + threadIdx.x;
    if (n < Nn) {
        float s = 0.f;
        #pragma unroll 4
        for (int k = 0; k < Dd; ++k) { float v = W[(size_t)n*Dd + k]; s += v*v; }
        nrm[n] = sqrtf(s);
    }
}

// ---------------- 2) cosine similarity + top-20 per row ----------------
__global__ void cosine_topk_k(const float* __restrict__ W, const float* __restrict__ nrm,
                              int* __restrict__ topk) {
    __shared__ float wrow[Dd];
    __shared__ float cosv[Nn];
    __shared__ float rmax[256];
    __shared__ int   ridx[256];
    const int i = blockIdx.x;          // row
    const int t = threadIdx.x;         // 0..255
    for (int k = t; k < Dd; k += 256) wrow[k] = W[(size_t)i*Dd + k];
    __syncthreads();
    const float ni = nrm[i];
    for (int c = t; c < Nn; c += 256) {
        const float* wc = W + (size_t)c*Dd;
        float dp = 0.f;
        #pragma unroll 8
        for (int k = 0; k < Dd; ++k) dp += wrow[k] * wc[k];
        cosv[c] = dp / (ni * nrm[c]);
    }
    __syncthreads();
    for (int sel = 0; sel < TOPK; ++sel) {
        float best = -1e30f; int bi = Nn;
        for (int c = t; c < Nn; c += 256) {
            float v = cosv[c];
            if (v > best || (v == best && c < bi)) { best = v; bi = c; }
        }
        rmax[t] = best; ridx[t] = bi;
        __syncthreads();
        for (int s = 128; s > 0; s >>= 1) {
            if (t < s) {
                float v2 = rmax[t+s]; int i2 = ridx[t+s];
                if (v2 > rmax[t] || (v2 == rmax[t] && i2 < ridx[t])) { rmax[t]=v2; ridx[t]=i2; }
            }
            __syncthreads();
        }
        if (t == 0) { topk[i*TOPK + sel] = ridx[0]; cosv[ridx[0]] = -1e30f; }
        __syncthreads();
    }
}

// ---------------- tiled WMMA GEMM: Out = A[M,K] @ Wb[Nout,K]^T ----------------
// Block = 8 waves, block tile 128(M) x 64(N). bf16 B panel (64 x 32 = 4KB) staged
// in LDS via GLOBAL_LOAD_ASYNC_TO_LDS_B128 (ASYNCcnt), shared by all 8 waves.
// Each wave: 1 A fragment + 4 B fragments -> 4 x v_wmma_f32_16x16x32_bf16 / K-step.
// ABF16: A is bf16 row-major (pure b128 loads, no cvt). STRIDED: A gathered fp32
// from data[b, k, n]. OBF16: store output as bf16. EP=1: fused bias + sigmoid.
template<int EP, int STRIDED, int ABF16, int OBF16>
__global__ __launch_bounds__(256)
void gemm_wmma_tiled(const void* __restrict__ Avoid, const __bf16* __restrict__ Wb,
                     const float* __restrict__ bias, void* __restrict__ Outv,
                     int M, int K, int Nout) {
    __shared__ __bf16 panelb[TILE_N][PANWB];
    const int nb   = Nout / TILE_N;
    const int m0   = (blockIdx.x / nb) * TILE_M;
    const int n0   = (blockIdx.x % nb) * TILE_N;
    const int w    = threadIdx.x >> 5;
    const int lane = threadIdx.x & 31;
    const int half = lane >> 4, l16 = lane & 15;
    const int tm   = m0 + w * 16;
    const int arow = tm + l16;

    const float*  Af = (const float*)Avoid;
    const __bf16* Ab = (const __bf16*)Avoid;
    const float*  Asg = nullptr;      // strided fp32 gather base
    const __bf16* Arb = nullptr;      // bf16 row base
    if (STRIDED) {
        const int b = arow / Nn, n = arow - b * Nn;
        Asg = Af + (size_t)b * F_IN * Nn + n;       // element k at Asg[k*Nn]
    } else if (ABF16) {
        Arb = Ab + (size_t)arow * K;
    }

    v8f acc[4] = {};

    for (int k0 = 0; k0 < K; k0 += 32) {
        // --- stage bf16 B panel: 64 rows x 32 bf16 (256 x b128 transfers, 1/thread) ---
        {
            const int r  = threadIdx.x >> 2;
            const int c8 = (threadIdx.x & 3) << 3;
            const __bf16* gp = Wb + (size_t)(n0 + r) * K + k0 + c8;
#if HAVE_ASYNC_LDS
            __builtin_amdgcn_global_load_async_to_lds_b128(
                (__attribute__((address_space(1))) v4i*)gp,
                (__attribute__((address_space(3))) v4i*)&panelb[r][c8], 0, 0);
#else
            *(uint4*)&panelb[r][c8] = *(const uint4*)gp;
#endif
        }
#if HAVE_ASYNC_LDS
        wait_async0();
#endif
        __syncthreads();

        // --- A fragment (16x32 bf16 layout per ISA) ---
        v16bf a;
        const int ka0 = k0 + (half ? 8 : 0);
        const int ka1 = k0 + (half ? 24 : 16);
        if (ABF16) {
            v8bf lo = *(const v8bf*)(Arb + ka0);    // 16B load, no cvt
            v8bf hi = *(const v8bf*)(Arb + ka1);
            #pragma unroll
            for (int e = 0; e < 8; ++e) { a[e] = lo[e]; a[8+e] = hi[e]; }
            if (k0 + 32 < K) __builtin_prefetch(Arb + k0 + 32, 0, 1);   // global_prefetch_b8
        } else if (STRIDED) {
            #pragma unroll
            for (int e = 0; e < 8; ++e) a[e]     = (__bf16)Asg[(size_t)(ka0 + e) * Nn];
            #pragma unroll
            for (int e = 0; e < 8; ++e) a[8 + e] = (__bf16)Asg[(size_t)(ka1 + e) * Nn];
        }

        // --- 4 B fragments from LDS panel, 4 WMMAs (A reused 4x) ---
        const int kb = half ? 8 : 0;
        #pragma unroll
        for (int nf = 0; nf < 4; ++nf) {
            const __bf16* lp = &panelb[nf * 16 + l16][kb];
            v8bf b0 = *(const v8bf*)(lp);
            v8bf b1 = *(const v8bf*)(lp + 16);
            v16bf bf;
            #pragma unroll
            for (int e = 0; e < 8; ++e) { bf[e] = b0[e]; bf[8+e] = b1[e]; }
            acc[nf] = __builtin_amdgcn_wmma_f32_16x16x32_bf16(false, a, false, bf,
                                                              (short)0, acc[nf], false, false);
        }
        __syncthreads();   // panel reused next iteration
    }

    // --- epilogue: 4 x (16x16) f32 tiles ---
    #pragma unroll
    for (int nf = 0; nf < 4; ++nf) {
        const int col = n0 + nf * 16 + l16;
        const float bs = (EP == 1) ? bias[col] : 0.f;
        #pragma unroll
        for (int v = 0; v < 8; ++v) {
            const int row = tm + v + (half ? 8 : 0);
            float val = acc[nf][v];
            if (EP == 1) val = 1.f / (1.f + expf(-(val + bs)));
            if (OBF16) ((__bf16*)Outv)[(size_t)row * Nout + col] = (__bf16)val;
            else       ((float*) Outv)[(size_t)row * Nout + col] = val;
        }
    }
}

// ---------------- per-node attention scalars: dot(row,u), dot(row,w) ----------------
__global__ void node_dots_k(const float* __restrict__ rows, const float* __restrict__ u,
                            const float* __restrict__ w, float* __restrict__ su,
                            float* __restrict__ sw, int nrows) {
    const int wave = blockIdx.x * (blockDim.x >> 5) + (threadIdx.x >> 5);
    const int lane = threadIdx.x & 31;
    if (wave >= nrows) return;
    const float* r = rows + (size_t)wave * Dd;
    float a = 0.f, b = 0.f;
    #pragma unroll
    for (int k = lane; k < Dd; k += 32) { float v = r[k]; a += v*u[k]; b += v*w[k]; }
    #pragma unroll
    for (int o = 16; o > 0; o >>= 1) { a += __shfl_xor(a, o, 32); b += __shfl_xor(b, o, 32); }
    if (lane == 0) { su[wave] = a; sw[wave] = b; }
}

// ---------------- attention softmax + aggregate + BN + ReLU + emb gate ----------------
// writes transposed gated output as bf16: gatedT[b, d, n]
__global__ void gdn_attn_agg_k(const float* __restrict__ xl, const float* __restrict__ emb,
                               const int* __restrict__ topk,
                               const float* __restrict__ s_i, const float* __restrict__ s_j,
                               const float* __restrict__ t_i, const float* __restrict__ t_j,
                               const float* __restrict__ gnn_bias, const float* __restrict__ gam,
                               const float* __restrict__ bet, __bf16* __restrict__ gatedT) {
    __shared__ float al[TOPK];
    __shared__ float aw[TOPK];
    __shared__ int   nb[TOPK];
    const int v = blockIdx.x;               // destination node, 0..B*N-1
    const int b = v / Nn, n = v - b*Nn;
    const int d = threadIdx.x;              // 0..D-1
    if (d < TOPK) {
        const int j = topk[n*TOPK + d];     // source node within batch
        float a = s_i[v] + t_i[n] + s_j[b*Nn + j] + t_j[j];
        a = (a > 0.f) ? a : 0.2f * a;       // leaky_relu(0.2)
        al[d] = a; nb[d] = b*Nn + j;
    }
    __syncthreads();
    if (d < TOPK) {
        float m = al[0];
        #pragma unroll
        for (int t = 1; t < TOPK; ++t) m = fmaxf(m, al[t]);
        float ssum = 0.f;
        #pragma unroll
        for (int t = 0; t < TOPK; ++t) ssum += expf(al[t] - m);
        aw[d] = expf(al[d] - m) / ssum;
    }
    __syncthreads();
    float acc = 0.f;
    #pragma unroll 4
    for (int t = 0; t < TOPK; ++t) acc += aw[t] * xl[(size_t)nb[t]*Dd + d];
    const float bnscale = 0.9999950000374997f;      // 1/sqrt(1+1e-5)
    float h = (acc + gnn_bias[d]) * (gam[d] * bnscale) + bet[d];
    h = fmaxf(h, 0.f);
    h *= emb[(size_t)n*Dd + d];
    gatedT[((size_t)b*Dd + d)*Nn + n] = (__bf16)h;
}

// ---------------- tail outputs: emb_weight copy + topk_idx (bit-cast int) ------------
__global__ void write_tail_k(const float* __restrict__ emb, const int* __restrict__ topk,
                             float* __restrict__ tail) {
    const int i = blockIdx.x * blockDim.x + threadIdx.x;
    if (i < Nn*Dd) tail[i] = emb[i];
    if (i < Nn*TOPK) ((int*)(tail + Nn*Dd))[i] = topk[i];
}

// ---------------- workspace layout ----------------
// fp32 region (float offsets)
#define OFS_NRM   ((size_t)0)                       // 512
#define OFS_TI    ((size_t)512)                     // 512
#define OFS_TJ    ((size_t)1024)                    // 512
#define OFS_TOPK  ((size_t)1536)                    // 10240 ints
#define OFS_SI    ((size_t)11776)                   // 32768
#define OFS_SJ    ((size_t)44544)                   // 32768
#define OFS_XL    ((size_t)77312)                   // 4194304
#define OFS_BF    (OFS_XL + (size_t)4194304)        // bf16 region starts here
// bf16 region (bf16 offsets from base)
#define BOF_LW    ((size_t)0)                       // 32768   lin_w bf16
#define BOF_W1    ((size_t)32768)                   // 262144  w1 bf16
#define BOF_W2    ((size_t)294912)                  // 262144  w2 bf16
#define BOF_GT    ((size_t)557056)                  // 4194304 gatedT bf16
#define BOF_H1    ((size_t)4751360)                 // 4194304 h1 bf16
// total ~17.1 MB fp32 + ~17.9 MB bf16 = ~35 MB

extern "C" void kernel_launch(void* const* d_in, const int* in_sizes, int n_in,
                              void* d_out, int out_size, void* d_ws, size_t ws_size,
                              hipStream_t stream) {
    const float* data     = (const float*)d_in[0];
    const float* emb      = (const float*)d_in[1];
    const float* lin_w    = (const float*)d_in[2];
    const float* att_i    = (const float*)d_in[3];
    const float* att_j    = (const float*)d_in[4];
    const float* att_em_i = (const float*)d_in[5];
    const float* att_em_j = (const float*)d_in[6];
    const float* gnn_bias = (const float*)d_in[7];
    const float* bn_gamma = (const float*)d_in[8];
    const float* bn_beta  = (const float*)d_in[9];
    const float* w1       = (const float*)d_in[10];
    const float* w2       = (const float*)d_in[11];
    const float* b2       = (const float*)d_in[12];

    float*  wsf    = (float*)d_ws;
    float*  nrm    = wsf + OFS_NRM;
    float*  t_i    = wsf + OFS_TI;
    float*  t_j    = wsf + OFS_TJ;
    int*    topk   = (int*)(wsf + OFS_TOPK);
    float*  s_i    = wsf + OFS_SI;
    float*  s_j    = wsf + OFS_SJ;
    float*  xl     = wsf + OFS_XL;
    __bf16* bfb    = (__bf16*)(wsf + OFS_BF);
    __bf16* lin_wb = bfb + BOF_LW;
    __bf16* w1b    = bfb + BOF_W1;
    __bf16* w2b    = bfb + BOF_W2;
    __bf16* gatedT = bfb + BOF_GT;
    __bf16* h1b    = bfb + BOF_H1;
    float*  out    = (float*)d_out;

    // 0) one-time weight conversions to bf16
    f2bf_k<<<(Dd*F_IN + 255)/256, 256, 0, stream>>>(lin_w, lin_wb, Dd*F_IN);
    f2bf_k<<<(HID*Nn + 255)/256, 256, 0, stream>>>(w1, w1b, HID*Nn);
    f2bf_k<<<(HID*HID + 255)/256, 256, 0, stream>>>(w2, w2b, HID*HID);
    // 1) norms
    row_norms_k<<<(Nn + 255)/256, 256, 0, stream>>>(emb, nrm);
    // 2) cosine top-k graph
    cosine_topk_k<<<Nn, 256, 0, stream>>>(emb, nrm, topk);
    // 3) xl = x @ lin_w^T : M=32768, K=256, Nout=128 -> 512 blocks
    {
        const int g = ((Bn*Nn)/TILE_M) * (Dd/TILE_N);
        gemm_wmma_tiled<0,1,0,0><<<g, 256, 0, stream>>>(data, lin_wb, nullptr, xl,
                                                        Bn*Nn, F_IN, Dd);
    }
    // 4) per-node scalars: s_i/s_j over xl rows; t_i/t_j over emb rows
    node_dots_k<<<(Bn*Nn + 7)/8, 256, 0, stream>>>(xl, att_i, att_j, s_i, s_j, Bn*Nn);
    node_dots_k<<<(Nn + 7)/8, 256, 0, stream>>>(emb, att_em_i, att_em_j, t_i, t_j, Nn);
    // 5) attention + aggregate + BN/ReLU + gate -> gatedT [B, D, N] (bf16)
    gdn_attn_agg_k<<<Bn*Nn, Dd, 0, stream>>>(xl, emb, topk, s_i, s_j, t_i, t_j,
                                             gnn_bias, bn_gamma, bn_beta, gatedT);
    // 6) h1 = gatedT @ w1^T : M=8192, K=512, Nout=512 -> 512 blocks (bf16 A, bf16 out)
    {
        const int g = ((Bn*Dd)/TILE_M) * (HID/TILE_N);
        gemm_wmma_tiled<0,0,1,1><<<g, 256, 0, stream>>>(gatedT, w1b, nullptr, h1b,
                                                        Bn*Dd, Nn, HID);
        // 7) out = sigmoid(h1 @ w2^T + b2) -> fp32 straight into d_out
        gemm_wmma_tiled<1,0,1,0><<<g, 256, 0, stream>>>(h1b, w2b, b2, out,
                                                        Bn*Dd, HID, HID);
    }
    // 8) tuple tail: emb_weight + topk_idx (bit-cast)
    write_tail_k<<<(Nn*Dd + 255)/256, 256, 0, stream>>>(emb, topk,
                                                        out + (size_t)Bn*Dd*HID);
}
